// MultiHeadAttention_62843961475635
// MI455X (gfx1250) — matrix operand order repaired and checked by reading, not compiled, not run on hardware
//
#include <hip/hip_runtime.h>
#include <hip/hip_bf16.h>
#include <stdint.h>

// ---------------- problem constants ----------------
#define D_MODEL 1024
#define NHEAD   16
#define DK      64
#define SEQ     2048
#define BATCH   2
#define MROWS   (BATCH * SEQ)   // 4096 rows for the projection GEMMs

// ---------------- CDNA5 WMMA types ----------------
typedef __attribute__((ext_vector_type(16))) __bf16 v16bf;
typedef __attribute__((ext_vector_type(8)))  float  v8f;

#define WMMA_BF16(a, b, c) \
  __builtin_amdgcn_wmma_f32_16x16x32_bf16(false, (a), false, (b), (short)0, (c), false, false)

// round-to-nearest-even f32 -> bf16
static __device__ __forceinline__ uint16_t f2bf(float x) {
  uint32_t u = __float_as_uint(x);
  u += 0x7FFFu + ((u >> 16) & 1u);
  return (uint16_t)(u >> 16);
}
static __device__ __forceinline__ uint32_t pk2bf(float a, float b) {
  return (uint32_t)f2bf(a) | ((uint32_t)f2bf(b) << 16);
}

union FragAB {
  v16bf    v;
  uint4    q[2];
  uint32_t w[8];
};

// A fragment 16x32 (MxK), row-major along K. rowPtr = &A[(m-row of this lane)][0].
// lane<16: K = {k..k+7, k+16..k+23}; lane>=16: shifted by 8. (ISA 7.12.2)
static __device__ __forceinline__ v16bf ldA(const uint16_t* rowPtr, int k, int h) {
  FragAB f;
  f.q[0] = *(const uint4*)(rowPtr + k + 8 * h);
  f.q[1] = *(const uint4*)(rowPtr + k + 16 + 8 * h);
  return f.v;
}
// B fragment 32x16 (KxN) taken from a row-major (N x K) matrix (NT GEMM):
// lane holds column n = lane&15 -> row n of W; lane<16: K = {k..k+7, k+16..k+23};
// lane>=16: shifted by 8 (same order as ldA) -> two 16B loads.
static __device__ __forceinline__ v16bf ldB(const uint16_t* rowPtr, int k, int h) {
  FragAB f;
  f.q[0] = *(const uint4*)(rowPtr + k + 8 * h);
  f.q[1] = *(const uint4*)(rowPtr + k + 16 + 8 * h);
  return f.v;
}

// ---------------- kernel 1: f32 -> bf16 ----------------
__global__ void cvt_f32_bf16(const float* __restrict__ src, uint16_t* __restrict__ dst, int n) {
  int i = blockIdx.x * blockDim.x + threadIdx.x;
  int stride = gridDim.x * blockDim.x;
  for (; i < n; i += stride) dst[i] = f2bf(src[i]);
}

// ---------------- shared GEMM core (NT, K = 1024, wave tile 16x64) ----------------
// Ping-pong double buffering (K-loop unrolled by 2): while WMMAs consume one
// buffer the other buffer's loads are in flight; buffers keep fixed registers
// so no copy churn.
static __device__ __forceinline__ void gemm_core_16x64(
    const uint16_t* __restrict__ aRow, const uint16_t* __restrict__ bRow,
    int h, v8f c[4]) {
  v16bf aP = ldA(aRow, 0, h);
  v16bf bP[4];
#pragma unroll
  for (int sub = 0; sub < 4; ++sub) bP[sub] = ldB(bRow + (size_t)sub * 16 * D_MODEL, 0, h);

#pragma unroll 1
  for (int k = 0; k < D_MODEL; k += 64) {
    // buffer Q <- fragments for k+32 (in flight during first WMMA group)
    v16bf aQ = ldA(aRow, k + 32, h);
    v16bf bQ[4];
#pragma unroll
    for (int sub = 0; sub < 4; ++sub) bQ[sub] = ldB(bRow + (size_t)sub * 16 * D_MODEL, k + 32, h);
    __builtin_prefetch(aRow + k + 192, 0, 3);   // global_prefetch_b8, ~3 steps ahead
    __builtin_prefetch(bRow + k + 192, 0, 3);
#pragma unroll
    for (int sub = 0; sub < 4; ++sub) c[sub] = WMMA_BF16(aP, bP[sub], c[sub]);
    // buffer P <- fragments for k+64 (in flight during second WMMA group)
    if (k + 64 < D_MODEL) {
      aP = ldA(aRow, k + 64, h);
#pragma unroll
      for (int sub = 0; sub < 4; ++sub) bP[sub] = ldB(bRow + (size_t)sub * 16 * D_MODEL, k + 64, h);
    }
#pragma unroll
    for (int sub = 0; sub < 4; ++sub) c[sub] = WMMA_BF16(aQ, bQ[sub], c[sub]);
  }
}

// ---------------- kernel 2: fused QKV projection ----------------
// Y[m,e] = sum_d X[m,d] * W[e,d] + bias[e]   (NT GEMM)
// z = 0: Q -> [B,H,S,dk]   z = 1: K -> [B,H,S,dk]   z = 2: V -> [B,H,dk,S] (transposed)
__global__ void __launch_bounds__(256) qkv_gemm(
    const uint16_t* __restrict__ X,
    const uint16_t* __restrict__ Wq, const uint16_t* __restrict__ Wk, const uint16_t* __restrict__ Wv,
    const float* __restrict__ bq, const float* __restrict__ bk, const float* __restrict__ bv,
    uint16_t* __restrict__ Q, uint16_t* __restrict__ Kt, uint16_t* __restrict__ Vt) {
  const int z = blockIdx.z;
  const uint16_t* W   = (z == 0) ? Wq : ((z == 1) ? Wk : Wv);
  const float*    bia = (z == 0) ? bq : ((z == 1) ? bk : bv);

  const int lane = threadIdx.x & 31;
  const int wave = threadIdx.x >> 5;
  const int h    = lane >> 4;
  const int l16  = lane & 15;
  const int m0   = blockIdx.x * 128 + wave * 16;
  const int n0   = blockIdx.y * 64;

  const uint16_t* aRow = X + (size_t)(m0 + l16) * D_MODEL;
  const uint16_t* bRow = W + (size_t)(n0 + l16) * D_MODEL;

  v8f c[4] = {};
  gemm_core_16x64(aRow, bRow, h, c);

#pragma unroll
  for (int sub = 0; sub < 4; ++sub) {
    const int   e  = n0 + sub * 16 + l16;
    const int   hh = e >> 6, d = e & 63;
    const float bs = bia[e];
#pragma unroll
    for (int i = 0; i < 8; ++i) {
      const int m = m0 + i + 8 * h;
      const int b = m >> 11, s = m & 2047;
      const uint16_t val = f2bf(c[sub][i] + bs);
      if (z == 2)
        Vt[(((size_t)b * NHEAD + hh) * DK + d) * SEQ + s] = val;   // V transposed
      else {
        uint16_t* O = (z == 0) ? Q : Kt;
        O[(((size_t)b * NHEAD + hh) * SEQ + s) * DK + d] = val;
      }
    }
  }
}

// ---------------- kernel 3: fused scores + softmax + P@V ----------------
// One block per (b, h, 16-row chunk). LDS: P = 16x2048 f32 (128KB) + O = 16x64 f32.
__global__ void __launch_bounds__(256) attn_kernel(
    const uint16_t* __restrict__ Q, const uint16_t* __restrict__ K,
    const uint16_t* __restrict__ Vt,
    float* __restrict__ probs, uint16_t* __restrict__ AO) {
  extern __shared__ float sm[];
  float* P = sm;                 // [16][SEQ]
  float* O = sm + 16 * SEQ;      // [16][64]

  const int bid   = blockIdx.x;
  const int chunk = bid & 127;           // S/16 = 128 chunks
  const int hh    = (bid >> 7) & 15;
  const int b     = bid >> 11;
  const int s0    = chunk * 16;
  const int lane  = threadIdx.x & 31;
  const int wave  = threadIdx.x >> 5;
  const int h     = lane >> 4;
  const int l16   = lane & 15;
  const size_t bh = (size_t)b * NHEAD + hh;

  for (int i = threadIdx.x; i < 16 * 64; i += 256) O[i] = 0.f;

  // ---- scores: S = (Q K^T) / sqrt(dk); 2 WMMA per 16x16 tile; ping-pong tiles ----
  const uint16_t* Qrow = Q + (bh * SEQ + s0 + l16) * DK;
  const v16bf a0 = ldA(Qrow, 0, h);
  const v16bf a1 = ldA(Qrow, 32, h);

  const uint16_t* Krow = K + (bh * SEQ + wave * 256 + l16) * DK;
  v16bf pb0 = ldB(Krow, 0, h);
  v16bf pb1 = ldB(Krow, 32, h);
#pragma unroll 1
  for (int tt = 0; tt < 16; tt += 2) {              // wave covers 256 columns
    // buffer Q <- tile tt+1 (in flight during tile tt WMMAs)
    const uint16_t* K1 = Krow + (size_t)(tt + 1) * 16 * DK;
    v16bf qb0 = ldB(K1, 0, h);
    v16bf qb1 = ldB(K1, 32, h);
    __builtin_prefetch(Krow + (size_t)(tt + 2) * 16 * DK, 0, 3);

    v8f c = {};
    c = WMMA_BF16(a0, pb0, c);
    c = WMMA_BF16(a1, pb1, c);
    const int t0 = wave * 256 + tt * 16;
#pragma unroll
    for (int i = 0; i < 8; ++i)
      P[(i + 8 * h) * SEQ + t0 + l16] = c[i] * 0.125f;   // 1/sqrt(64)

    // buffer P <- tile tt+2 (in flight during tile tt+1 WMMAs)
    if (tt + 2 < 16) {
      const uint16_t* K2 = Krow + (size_t)(tt + 2) * 16 * DK;
      pb0 = ldB(K2, 0, h);
      pb1 = ldB(K2, 32, h);
    }
    v8f c2 = {};
    c2 = WMMA_BF16(a0, qb0, c2);
    c2 = WMMA_BF16(a1, qb1, c2);
#pragma unroll
    for (int i = 0; i < 8; ++i)
      P[(i + 8 * h) * SEQ + t0 + 16 + l16] = c2[i] * 0.125f;
  }
  __syncthreads();

  // ---- softmax: wave handles rows 2w, 2w+1 ----
  for (int rr = 0; rr < 2; ++rr) {
    const int r = wave * 2 + rr;
    float* row = P + r * SEQ;
    float mx = -3.4e38f;
    for (int j = lane; j < SEQ; j += 32) mx = fmaxf(mx, row[j]);
    for (int off = 16; off; off >>= 1) mx = fmaxf(mx, __shfl_xor(mx, off, 32));
    float sum = 0.f;
    for (int j = lane; j < SEQ; j += 32) { float e = __expf(row[j] - mx); row[j] = e; sum += e; }
    for (int off = 16; off; off >>= 1) sum += __shfl_xor(sum, off, 32);
    const float inv = 1.f / sum;
    float* gout = probs + (bh * SEQ + s0 + r) * SEQ;
    for (int j = lane; j < SEQ; j += 32) { float p = row[j] * inv; row[j] = p; gout[j] = p; }
  }
  __syncthreads();

  // ---- P @ V: O[16x64] over K = 2048; 4 d-subtiles x 2 K-halves across 8 waves.
  // Ping-pong V fragments; the LDS bf16-pack VALU work overlaps the loads.
  {
    const int nsub = wave & 3, khalf = wave >> 2;
    const int d0 = nsub * 16;
    const int kbase = khalf * 1024;
    const uint16_t* Vrow = Vt + (bh * DK + d0 + l16) * SEQ;
    const float* prowB = P + l16 * SEQ + 8 * h;
    v8f c = {};
    v16bf vp = ldB(Vrow, kbase, h);
#pragma unroll 1
    for (int ks = 0; ks < 32; ks += 2) {
      const int k = kbase + ks * 32;
      v16bf vq = ldB(Vrow, k + 32, h);          // in flight during pack+WMMA of k
      FragAB fa;
      const float* prow = prowB + k;
#pragma unroll
      for (int j = 0; j < 4; ++j) {
        float2 f0 = *(const float2*)(prow + 2 * j);
        fa.w[j] = pk2bf(f0.x, f0.y);
        float2 f1 = *(const float2*)(prow + 16 + 2 * j);
        fa.w[4 + j] = pk2bf(f1.x, f1.y);
      }
      c = WMMA_BF16(fa.v, vp, c);
      if (ks + 2 < 32) vp = ldB(Vrow, k + 64, h);  // in flight during pack+WMMA of k+32
      FragAB fb;
      const float* prow2 = prowB + k + 32;
#pragma unroll
      for (int j = 0; j < 4; ++j) {
        float2 f0 = *(const float2*)(prow2 + 2 * j);
        fb.w[j] = pk2bf(f0.x, f0.y);
        float2 f1 = *(const float2*)(prow2 + 16 + 2 * j);
        fb.w[4 + j] = pk2bf(f1.x, f1.y);
      }
      c = WMMA_BF16(fb.v, vq, c);
    }
#pragma unroll
    for (int i = 0; i < 8; ++i)
      atomicAdd(&O[(i + 8 * h) * 64 + d0 + l16], c[i]);  // combine the 2 K-halves
  }
  __syncthreads();

  // ---- write attn_out (bf16, [B,S,D] with e = h*64+d) ----
  for (int i = threadIdx.x; i < 16 * 64; i += 256) {
    const int m = i >> 6, d = i & 63;
    AO[((size_t)b * SEQ + s0 + m) * D_MODEL + hh * DK + d] = f2bf(O[i]);
  }
}

// ---------------- kernel 4: output projection (fp32 out) ----------------
__global__ void __launch_bounds__(256) out_gemm(
    const uint16_t* __restrict__ A, const uint16_t* __restrict__ W,
    const float* __restrict__ bias, float* __restrict__ Out) {
  const int lane = threadIdx.x & 31;
  const int wave = threadIdx.x >> 5;
  const int h    = lane >> 4;
  const int l16  = lane & 15;
  const int m0   = blockIdx.x * 128 + wave * 16;
  const int n0   = blockIdx.y * 64;

  const uint16_t* aRow = A + (size_t)(m0 + l16) * D_MODEL;
  const uint16_t* bRow = W + (size_t)(n0 + l16) * D_MODEL;

  v8f c[4] = {};
  gemm_core_16x64(aRow, bRow, h, c);

#pragma unroll
  for (int sub = 0; sub < 4; ++sub) {
    const int   e  = n0 + sub * 16 + l16;
    const float bs = bias[e];
#pragma unroll
    for (int i = 0; i < 8; ++i) {
      const int m = m0 + i + 8 * h;
      Out[(size_t)m * D_MODEL + e] = c[sub][i] + bs;
    }
  }
}

// ---------------- host launch ----------------
extern "C" void kernel_launch(void* const* d_in, const int* in_sizes, int n_in,
                              void* d_out, int out_size, void* d_ws, size_t ws_size,
                              hipStream_t stream) {
  const float* x  = (const float*)d_in[0];
  const float* Wq = (const float*)d_in[1];
  const float* bq = (const float*)d_in[2];
  const float* Wk = (const float*)d_in[3];
  const float* bk = (const float*)d_in[4];
  const float* Wv = (const float*)d_in[5];
  const float* bv = (const float*)d_in[6];
  const float* Wo = (const float*)d_in[7];
  const float* bo = (const float*)d_in[8];

  char* ws = (char*)d_ws;
  const size_t MB = 1ull << 20;
  uint16_t* xb  = (uint16_t*)(ws + 0);        //  8 MB: x bf16
  uint16_t* wqb = (uint16_t*)(ws + 8  * MB);  //  2 MB each: weights bf16
  uint16_t* wkb = (uint16_t*)(ws + 10 * MB);
  uint16_t* wvb = (uint16_t*)(ws + 12 * MB);
  uint16_t* wob = (uint16_t*)(ws + 14 * MB);
  uint16_t* Qb  = (uint16_t*)(ws + 16 * MB);  //  8 MB: Q  [B,H,S,dk]
  uint16_t* Kb  = (uint16_t*)(ws + 24 * MB);  //  8 MB: K  [B,H,S,dk]
  uint16_t* Vtb = (uint16_t*)(ws + 32 * MB);  //  8 MB: V^T [B,H,dk,S]
  uint16_t* AOb = (uint16_t*)(ws + 40 * MB);  //  8 MB: attn_out bf16 [B,S,D]

  float* out   = (float*)d_out;
  float* probs = out + (size_t)MROWS * D_MODEL;  // attn_weights block of the tuple

  cvt_f32_bf16<<<1024, 256, 0, stream>>>(x,  xb,  MROWS * D_MODEL);
  cvt_f32_bf16<<<512,  256, 0, stream>>>(Wq, wqb, D_MODEL * D_MODEL);
  cvt_f32_bf16<<<512,  256, 0, stream>>>(Wk, wkb, D_MODEL * D_MODEL);
  cvt_f32_bf16<<<512,  256, 0, stream>>>(Wv, wvb, D_MODEL * D_MODEL);
  cvt_f32_bf16<<<512,  256, 0, stream>>>(Wo, wob, D_MODEL * D_MODEL);

  qkv_gemm<<<dim3(32, 16, 3), 256, 0, stream>>>(xb, wqb, wkb, wvb, bq, bk, bv, Qb, Kb, Vtb);

  const size_t attn_lds = (size_t)16 * SEQ * 4 + 16 * 64 * 4;  // 132 KB
  attn_kernel<<<BATCH * NHEAD * (SEQ / 16), 256, attn_lds, stream>>>(Qb, Kb, Vtb, probs, AOb);

  out_gemm<<<dim3(32, 16), 256, 0, stream>>>(AOb, wob, bo, out);
}
